// Graph_OurConvNet_18966575579182
// MI455X (gfx1250) — matrix operand order repaired and checked by reading, not compiled, not run on hardware
//
#include <hip/hip_runtime.h>
#include <hip/hip_bf16.h>

#define N_NODES 100000
#define N_EDGES 800000
#define D 128
#define N_CLUST 8

typedef __attribute__((ext_vector_type(16))) __bf16 v16bf;
typedef __attribute__((ext_vector_type(8)))  float  v8f;

// Convert two runs of 8 contiguous f32 into a 16-element bf16 fragment.
__device__ __forceinline__ v16bf cvt2x8(const float* __restrict__ p0,
                                        const float* __restrict__ p1) {
  const float4* a = (const float4*)p0;
  const float4* b = (const float4*)p1;
  float4 x0 = a[0], x1 = a[1], y0 = b[0], y1 = b[1];
  v16bf r;
  r[0]  = (__bf16)x0.x; r[1]  = (__bf16)x0.y; r[2]  = (__bf16)x0.z; r[3]  = (__bf16)x0.w;
  r[4]  = (__bf16)x1.x; r[5]  = (__bf16)x1.y; r[6]  = (__bf16)x1.z; r[7]  = (__bf16)x1.w;
  r[8]  = (__bf16)y0.x; r[9]  = (__bf16)y0.y; r[10] = (__bf16)y0.z; r[11] = (__bf16)y0.w;
  r[12] = (__bf16)y1.x; r[13] = (__bf16)y1.y; r[14] = (__bf16)y1.z; r[15] = (__bf16)y1.w;
  return r;
}

// One wave computes Y[rt*16 .. rt*16+15, 0..127] = X_tile @ W^T (+bias), bf16 WMMA.
// A layout (16x32 bf16): lanes 0-15 row=lane, K=0-7 (v0-3) & 16-23 (v4-7);
//                        lanes 16-31 row=lane-16, K=8-15 & 24-31.
// B layout (32x16 bf16): lane = column (mod 16); lanes 0-15 K=0-7 & 16-23, lanes 16-31 K=8-15 & 24-31.
__device__ __forceinline__ void gemm_tile(const float* __restrict__ X,
                                          const float* __restrict__ W,
                                          const float* __restrict__ bias,
                                          float* __restrict__ Y, int rt) {
  const int lane = threadIdx.x & 31;
  const int mrow = lane & 15;
  const int hi   = (lane >> 4) & 1;
  int gr = rt * 16 + mrow;
  if (gr >= N_NODES) gr = N_NODES - 1;     // clamp (stores are guarded)
  const int kbA = hi ? 8 : 0;
  const int kbB = hi ? 8 : 0;
  const float* xrow = X + (size_t)gr * D;

  v8f acc[8] = {};

#pragma unroll
  for (int c = 0; c < 4; ++c) {            // K chunks of 32
    const float* ap = xrow + c * 32 + kbA;
    v16bf af = cvt2x8(ap, ap + 16);
#pragma unroll
    for (int t = 0; t < 8; ++t) {          // N tiles of 16
      const float* bp = W + (size_t)(t * 16 + mrow) * D + c * 32 + kbB;
      v16bf bf = cvt2x8(bp, bp + 16);
      acc[t] = __builtin_amdgcn_wmma_f32_16x16x32_bf16(
          false, af, false, bf, (short)0, acc[t], false, false);
    }
  }

  // D layout: lane l, vgpr r -> (M = r + 8*hi, N = l&15)
#pragma unroll
  for (int t = 0; t < 8; ++t) {
    const int colg = t * 16 + mrow;
    const float bb = bias ? bias[colg] : 0.0f;
#pragma unroll
    for (int r = 0; r < 8; ++r) {
      const int row = rt * 16 + r + hi * 8;
      if (row < N_NODES) Y[(size_t)row * D + colg] = acc[t][r] + bb;
    }
  }
}

// 4 waves/block, one weight matrix per wave, same 16-row tile.
__global__ void __launch_bounds__(128)
k_gemm4(const float* __restrict__ X,
        const float* __restrict__ W0, const float* __restrict__ W1,
        const float* __restrict__ W2, const float* __restrict__ W3,
        const float* __restrict__ bias3,
        float* __restrict__ Y0, float* __restrict__ Y1,
        float* __restrict__ Y2, float* __restrict__ Y3) {
  const int wv = threadIdx.x >> 5;
  const int rt = blockIdx.x;
  if (wv == 0)      gemm_tile(X, W0, nullptr, Y0, rt);
  else if (wv == 1) gemm_tile(X, W1, nullptr, Y1, rt);
  else if (wv == 2) gemm_tile(X, W2, nullptr, Y2, rt);
  else              gemm_tile(X, W3, bias3,  Y3, rt);
}

// 4 waves/block, each wave a different row tile (single weight).
__global__ void __launch_bounds__(128)
k_gemm1(const float* __restrict__ X, const float* __restrict__ W,
        float* __restrict__ Y) {
  const int rt = blockIdx.x * 4 + (threadIdx.x >> 5);
  if (rt * 16 >= N_NODES) return;
  gemm_tile(X, W, nullptr, Y, rt);
}

// x = emb[node_ids], float4-vectorized.
__global__ void k_gather(const float* __restrict__ emb, const int* __restrict__ ids,
                         float* __restrict__ x) {
  int gid = blockIdx.x * blockDim.x + threadIdx.x;        // float4 index
  if (gid >= N_NODES * (D / 4)) return;
  int i = gid >> 5, q = gid & 31;
  ((float4*)x)[gid] = ((const float4*)emb)[(size_t)ids[i] * (D / 4) + q];
}

// One wave per edge: acc[end] += sigmoid(w*Vix[end]+w*Vjx[start]+bv) * (w*Ujx[start])
__global__ void k_edge(const float* __restrict__ vix, const float* __restrict__ vjx,
                       const float* __restrict__ ujx, const float* __restrict__ bv,
                       const int* __restrict__ sidx, const int* __restrict__ eidx,
                       const int* __restrict__ ew, float* __restrict__ acc) {
  int gid  = blockIdx.x * blockDim.x + threadIdx.x;
  int e    = gid >> 5;
  int lane = gid & 31;
  if (e >= N_EDGES) return;
  int s = sidx[e], t = eidx[e];
  float w = (float)ew[e];
  float4 a   = ((const float4*)(vix + (size_t)t * D))[lane];
  float4 b   = ((const float4*)(vjx + (size_t)s * D))[lane];
  float4 u   = ((const float4*)(ujx + (size_t)s * D))[lane];
  float4 bvv = ((const float4*)bv)[lane];
  float* dst = acc + (size_t)t * D + lane * 4;
  float z, g;
  z = w * a.x + w * b.x + bvv.x; g = 1.0f / (1.0f + __expf(-z)); atomicAdd(dst + 0, g * (w * u.x));
  z = w * a.y + w * b.y + bvv.y; g = 1.0f / (1.0f + __expf(-z)); atomicAdd(dst + 1, g * (w * u.y));
  z = w * a.z + w * b.z + bvv.z; g = 1.0f / (1.0f + __expf(-z)); atomicAdd(dst + 2, g * (w * u.z));
  z = w * a.w + w * b.w + bvv.w; g = 1.0f / (1.0f + __expf(-z)); atomicAdd(dst + 3, g * (w * u.w));
}

__global__ void k_zero(float* __restrict__ p, int n) {
  int gid = blockIdx.x * blockDim.x + threadIdx.x;
  if (gid < n) p[gid] = 0.0f;
}

// Per-channel sum / sumsq partials; thread = channel -> coalesced row reads.
__global__ void __launch_bounds__(128)
k_reduce(const float* __restrict__ x, float* __restrict__ stats, int rpb) {
  int d  = threadIdx.x;
  int r0 = blockIdx.x * rpb;
  float s = 0.0f, s2 = 0.0f;
  for (int r = 0; r < rpb; ++r) {
    int row = r0 + r;
    if (row < N_NODES) {
      float v = x[(size_t)row * D + d];
      s += v; s2 += v * v;
    }
  }
  atomicAdd(&stats[d], s);
  atomicAdd(&stats[D + d], s2);
}

__global__ void __launch_bounds__(128)
k_bnstats(const float* __restrict__ stats, const float* __restrict__ g,
          const float* __restrict__ b, float* __restrict__ scsh) {
  int d = threadIdx.x;
  const float invN = 1.0f / (float)N_NODES;
  float m   = stats[d] * invN;
  float var = stats[D + d] * invN - m * m;      // biased var, matches jnp.var
  float sc  = g[d] * rsqrtf(var + 1e-5f);
  scsh[d]     = sc;
  scsh[D + d] = b[d] - m * sc;
}

__global__ void k_apply1(const float* __restrict__ x, const float* __restrict__ scsh,
                         float* __restrict__ y) {
  int gid = blockIdx.x * blockDim.x + threadIdx.x;        // float4 index
  if (gid >= N_NODES * (D / 4)) return;
  int q = gid & (D / 4 - 1);
  float4 v  = ((const float4*)x)[gid];
  float4 sc = ((const float4*)scsh)[q];
  float4 sh = ((const float4*)(scsh + D))[q];
  float4 o;
  o.x = fmaxf(sc.x * v.x + sh.x, 0.0f);
  o.y = fmaxf(sc.y * v.y + sh.y, 0.0f);
  o.z = fmaxf(sc.z * v.z + sh.z, 0.0f);
  o.w = fmaxf(sc.w * v.w + sh.w, 0.0f);
  ((float4*)y)[gid] = o;
}

__global__ void k_apply2(const float* __restrict__ x, const float* __restrict__ scsh,
                         const float* __restrict__ res, float* __restrict__ y) {
  int gid = blockIdx.x * blockDim.x + threadIdx.x;
  if (gid >= N_NODES * (D / 4)) return;
  int q = gid & (D / 4 - 1);
  float4 v  = ((const float4*)x)[gid];
  float4 rr = ((const float4*)res)[gid];
  float4 sc = ((const float4*)scsh)[q];
  float4 sh = ((const float4*)(scsh + D))[q];
  float4 o;
  o.x = fmaxf(sc.x * v.x + sh.x + rr.x, 0.0f);
  o.y = fmaxf(sc.y * v.y + sh.y + rr.y, 0.0f);
  o.z = fmaxf(sc.z * v.z + sh.z + rr.z, 0.0f);
  o.w = fmaxf(sc.w * v.w + sh.w + rr.w, 0.0f);
  ((float4*)y)[gid] = o;
}

// out[i*8+c] = x[i] . fc_w[c] + fc_b[c]  (flat == reference's reshape(-1, N))
__global__ void k_fc(const float* __restrict__ X, const float* __restrict__ Wf,
                     const float* __restrict__ bf, float* __restrict__ out) {
  int gid = blockIdx.x * blockDim.x + threadIdx.x;
  if (gid >= N_NODES * N_CLUST) return;
  int i = gid >> 3, c = gid & 7;
  const float4* xr = (const float4*)(X + (size_t)i * D);
  const float4* wr = (const float4*)(Wf + (size_t)c * D);
  float s = bf[c];
#pragma unroll 8
  for (int k = 0; k < D / 4; ++k) {
    float4 a = xr[k], w4 = wr[k];
    s += a.x * w4.x + a.y * w4.y + a.z * w4.z + a.w * w4.w;
  }
  out[gid] = s;
}

extern "C" void kernel_launch(void* const* d_in, const int* in_sizes, int n_in,
                              void* d_out, int out_size, void* d_ws, size_t ws_size,
                              hipStream_t stream) {
  const int*   node_ids  = (const int*)d_in[0];
  const int*   start_idx = (const int*)d_in[1];
  const int*   end_idx   = (const int*)d_in[2];
  const int*   edge_w    = (const int*)d_in[3];
  const float* emb   = (const float*)d_in[4];
  const float* Ui1   = (const float*)d_in[5];
  const float* Uj1   = (const float*)d_in[6];
  const float* Vi1   = (const float*)d_in[7];
  const float* Vj1   = (const float*)d_in[8];
  const float* bu1   = (const float*)d_in[9];
  const float* bv1   = (const float*)d_in[10];
  const float* Ui2   = (const float*)d_in[11];
  const float* Uj2   = (const float*)d_in[12];
  const float* Vi2   = (const float*)d_in[13];
  const float* Vj2   = (const float*)d_in[14];
  const float* bu2   = (const float*)d_in[15];
  const float* bv2   = (const float*)d_in[16];
  const float* bn1_g = (const float*)d_in[17];
  const float* bn1_b = (const float*)d_in[18];
  const float* bn2_g = (const float*)d_in[19];
  const float* bn2_b = (const float*)d_in[20];
  const float* Rw    = (const float*)d_in[21];
  const float* fc_w  = (const float*)d_in[22];
  const float* fc_b  = (const float*)d_in[23];

  const size_t nd = (size_t)N_NODES * D;
  float* bx    = (float*)d_ws;       // layer input / final features
  float* bh    = bx + nd;            // post-BN1 hidden
  float* ba    = bh + nd;            // Vi*x
  float* bb    = ba + nd;            // Vj*x
  float* bc    = bb + nd;            // Uj*x
  float* bacc  = bc + nd;            // Ui*x + bu, then += segment sum
  float* bres  = bacc + nd;          // xin @ Rw^T
  float* stats = bres + nd;          // 256 floats: sum | sumsq
  float* scsh  = stats + 256;        // 256 floats: scale | shift

  const int gGath  = N_NODES * (D / 4) / 256;          // 12500
  const int gGemm  = N_NODES / 16;                     // 6250
  const int gGemm1 = (gGemm + 3) / 4;                  // 1563
  const int gEdge  = N_EDGES * 32 / 256;               // 100000
  const int rpb    = (N_NODES + 511) / 512;            // 196
  const int gFc    = (N_NODES * N_CLUST + 255) / 256;  // 3125

  k_gather<<<gGath, 256, 0, stream>>>(emb, node_ids, bx);

  for (int l = 0; l < 2; ++l) {
    const size_t wo = (size_t)l * D * D;
    const size_t bo = (size_t)l * D;

    // ---- agg block 1: h = Ui1*x + bu1 + segsum(sigmoid(...) * w*Uj1x[start]) ----
    k_gemm4<<<gGemm, 128, 0, stream>>>(bx, Vi1 + wo, Vj1 + wo, Uj1 + wo, Ui1 + wo,
                                       bu1 + bo, ba, bb, bc, bacc);
    k_edge<<<gEdge, 256, 0, stream>>>(ba, bb, bc, bv1 + bo, start_idx, end_idx,
                                      edge_w, bacc);
    k_zero<<<1, 256, 0, stream>>>(stats, 256);
    k_reduce<<<512, 128, 0, stream>>>(bacc, stats, rpb);
    k_bnstats<<<1, 128, 0, stream>>>(stats, bn1_g + bo, bn1_b + bo, scsh);
    k_apply1<<<gGath, 256, 0, stream>>>(bacc, scsh, bh);

    // ---- residual from layer input ----
    k_gemm1<<<gGemm1, 128, 0, stream>>>(bx, Rw + wo, bres);

    // ---- agg block 2 on h ----
    k_gemm4<<<gGemm, 128, 0, stream>>>(bh, Vi2 + wo, Vj2 + wo, Uj2 + wo, Ui2 + wo,
                                       bu2 + bo, ba, bb, bc, bacc);
    k_edge<<<gEdge, 256, 0, stream>>>(ba, bb, bc, bv2 + bo, start_idx, end_idx,
                                      edge_w, bacc);
    k_zero<<<1, 256, 0, stream>>>(stats, 256);
    k_reduce<<<512, 128, 0, stream>>>(bacc, stats, rpb);
    k_bnstats<<<1, 128, 0, stream>>>(stats, bn2_g + bo, bn2_b + bo, scsh);
    k_apply2<<<gGath, 256, 0, stream>>>(bacc, scsh, bres, bx);
  }

  k_fc<<<gFc, 256, 0, stream>>>(bx, fc_w, fc_b, (float*)d_out);
}
